// TRCategorical_73340861547014
// MI455X (gfx1250) — matrix operand order repaired and checked by reading, not compiled, not run on hardware
//
#include <hip/hip_runtime.h>

// ---------------------------------------------------------------------------
// Types for CDNA5 WMMA (wave32) + TDM descriptors
// ---------------------------------------------------------------------------
typedef __attribute__((ext_vector_type(16))) __bf16 v16bf;
typedef __attribute__((ext_vector_type(8)))  __bf16 v8bf;
typedef __attribute__((ext_vector_type(8)))  float  v8f;
typedef __attribute__((ext_vector_type(4)))  unsigned int u32x4;
typedef __attribute__((ext_vector_type(8)))  int    i32x8;
typedef __attribute__((ext_vector_type(4)))  int    i32x4;

#define DEV __device__ __forceinline__

static constexpr int KMODES = 8;
static constexpr int D      = 1024;   // categories per mode
static constexpr int R      = 64;     // TR rank
static constexpr int RR     = R * R;  // 4096
static constexpr int NDRAWS = 2048;
static constexpr int KC     = 64;     // GEMM k-chunk staged per TDM transfer

DEV float softplus_f(float x) {
  return fmaxf(x, 0.0f) + log1pf(__expf(-fabsf(x)));   // stable softplus
}

DEV unsigned hash3(unsigned a, unsigned b, unsigned c) {
  unsigned x = a * 0x9E3779B9u + b * 0x85EBCA6Bu + c * 0xC2B2AE35u + 0x27D4EB2Fu;
  x ^= x >> 16; x *= 0x7FEB352Du;
  x ^= x >> 15; x *= 0x846CA68Bu;
  x ^= x >> 16;
  return x;
}

// ---------------------------------------------------------------------------
// WMMA fragment helpers (bf16 16x16x32, contiguous k order, same for A and B)
// A: 16(M) x 32(K) row-major, leading dim `ld`:
//   lane<16 : row=lane,    K = k0+[0..15]
//   lane>=16: row=lane-16, K = k0+[16..31]
// B: 32(K) x 16(N), loaded from transposed (N-major) storage, one 32B load:
//   lane<16 : col=lane,    K = k0+[0..15]
//   lane>=16: col=lane-16, K = k0+[16..31]
// ---------------------------------------------------------------------------
DEV v16bf load_a_frag(const __bf16* base, int ld, int lane, int k0) {
  const int r    = lane & 15;
  const int koff = k0 + ((lane & 16) ? 16 : 0);
  const __bf16* p = base + (size_t)r * ld + koff;
  v8bf lo = *(const v8bf*)(p);
  v8bf hi = *(const v8bf*)(p + 8);
  v16bf f;
#pragma unroll
  for (int i = 0; i < 8; ++i) { f[i] = lo[i]; f[8 + i] = hi[i]; }
  return f;
}

DEV v16bf load_b_fragT(const __bf16* baseT, int ld, int lane, int k0) {
  const int n    = lane & 15;
  const int koff = k0 + ((lane & 16) ? 16 : 0);
  return *(const v16bf*)(baseT + (size_t)n * ld + koff);
}

DEV v8f wmma_bf16(v16bf a, v16bf b, v8f c) {
  return __builtin_amdgcn_wmma_f32_16x16x32_bf16(false, a, false, b,
                                                 (short)0, c, false, false);
}

// ---------------------------------------------------------------------------
// TDM: 2D tile (bf16) global -> LDS. D# per CDNA5 ISA.
//   group0: [1:0]=count=1 | [63:32]=lds_addr | [120:64]=global_addr | type=2
//   group1: data_size=1 (2B) | tensor_dim0/1 | tile_dim0/1 | dim0 stride
//   groups 2/3 unused (2D tensor) -> zero.
// ---------------------------------------------------------------------------
DEV void tdm_load_2d_bf16(unsigned lds_off, unsigned long long gaddr,
                          unsigned tile_d0, unsigned tile_d1,
                          unsigned tensor_d0, unsigned tensor_d1,
                          unsigned stride_elems) {
  u32x4 g0;
  g0[0] = 1u;                                                    // count=1
  g0[1] = lds_off;                                               // lds_addr
  g0[2] = (unsigned)gaddr;                                       // gaddr lo32
  g0[3] = (unsigned)((gaddr >> 32) & 0x01FFFFFFull) | (2u << 30);// gaddr hi + type=2
  i32x8 g1;
  g1[0] = (int)(1u << 16);                                       // data_size=2B
  g1[1] = (int)((tensor_d0 & 0xFFFFu) << 16);                    // tensor_dim0 lo16
  g1[2] = (int)((tensor_d0 >> 16) | ((tensor_d1 & 0xFFFFu) << 16));
  g1[3] = (int)((tensor_d1 >> 16) | (tile_d0 << 16));            // tile_dim0
  g1[4] = (int)(tile_d1 & 0xFFFFu);                              // tile_dim1, dim2=0
  g1[5] = (int)stride_elems;                                     // dim0 stride lo32
  g1[6] = 0;
  g1[7] = 0;
  i32x4 z4 = {0, 0, 0, 0};
#if defined(__clang_major__) && (__clang_major__ >= 23)
  i32x8 z8 = {0, 0, 0, 0, 0, 0, 0, 0};
  __builtin_amdgcn_tensor_load_to_lds(g0, g1, z4, z4, z8, 0);
#else
  __builtin_amdgcn_tensor_load_to_lds(g0, g1, z4, z4, 0);
#endif
}

// ---------------------------------------------------------------------------
// K1: S_i[rr] = sum_n softplus(core_i[n, rr])
// ---------------------------------------------------------------------------
__global__ __launch_bounds__(256)
void sum_core_kernel(const float* __restrict__ src, float* __restrict__ S) {
  int rr = blockIdx.x * 256 + threadIdx.x;
  float s = 0.0f;
  for (int n = 0; n < D; ++n) s += softplus_f(src[(size_t)n * RR + rr]);
  S[rr] = s;
}

// ---------------------------------------------------------------------------
// K2: heads chain. headT[j][n*64+m] = (S0..S_{j-1})[m][n] (bf16), head0 = I.
//     Also logTraceNorm = log(trace(S0..S7)).
// ---------------------------------------------------------------------------
__global__ __launch_bounds__(256)
void heads_kernel(const float* __restrict__ S,
                  __bf16* __restrict__ headT,
                  float* __restrict__ logTraceNorm) {
  __shared__ float A[RR], Bm[RR], C[RR];
  const int t = threadIdx.x;
  for (int i = t; i < RR; i += 256) {
    float v = ((i >> 6) == (i & 63)) ? 1.0f : 0.0f;
    C[i] = v;
    headT[i] = (__bf16)v;
  }
  __syncthreads();
  for (int j = 1; j < KMODES; ++j) {
    for (int i = t; i < RR; i += 256) { A[i] = C[i]; Bm[i] = S[(size_t)(j - 1) * RR + i]; }
    __syncthreads();
    for (int i = t; i < RR; i += 256) {
      int m = i >> 6, n = i & 63;
      float s = 0.0f;
      for (int k = 0; k < R; ++k) s += A[m * R + k] * Bm[k * R + n];
      C[i] = s;
    }
    __syncthreads();
    for (int i = t; i < RR; i += 256) {
      int m = i >> 6, n = i & 63;
      headT[(size_t)j * RR + n * R + m] = (__bf16)C[i];
    }
    __syncthreads();
  }
  for (int i = t; i < RR; i += 256) Bm[i] = S[(size_t)7 * RR + i];
  __syncthreads();
  float part = 0.0f;
  for (int i = t; i < RR; i += 256) {
    int a = i >> 6, k = i & 63;
    part += C[a * R + k] * Bm[k * R + a];
  }
  A[t] = part;
  __syncthreads();
  for (int s2 = 128; s2 > 0; s2 >>= 1) {
    if (t < s2) A[t] += A[t + s2];
    __syncthreads();
  }
  if (t == 0) *logTraceNorm = __logf(fmaxf(A[0], 1e-30f));
}

// ---------------------------------------------------------------------------
// K3: coreBf[i] = bf16(softplus(src[i]))
// ---------------------------------------------------------------------------
__global__ __launch_bounds__(256)
void convert_core_kernel(const float* __restrict__ src, __bf16* __restrict__ dst, int n) {
  int i = blockIdx.x * 256 + threadIdx.x;
  if (i < n) dst[i] = (__bf16)softplus_f(src[i]);
}

// ---------------------------------------------------------------------------
// K4: first-step marginal weights  w[n] = sum_k coreflat7[n,k] * headT7[k]
// ---------------------------------------------------------------------------
__global__ __launch_bounds__(256)
void first_weights_kernel(const __bf16* __restrict__ coreflat,
                          const __bf16* __restrict__ headT7,
                          float* __restrict__ w) {
  int n = blockIdx.x * 256 + threadIdx.x;
  const __bf16* row = coreflat + (size_t)n * RR;
  float s = 0.0f;
  for (int k = 0; k < RR; ++k) s += (float)row[k] * (float)headT7[k];
  w[n] = s;
}

// ---------------------------------------------------------------------------
// K5: big GEMM  W[b, n] = sum_k Mflat[b, k] * coreflat[n, k]
//     TDM double-buffered LDS staging: per chunk, wave 0 DMAs A(128xKC) and
//     B(64xKC) tiles; all 8 waves run WMMA from LDS. B fetched once per
//     block (was 8x redundant global traffic). tensorcnt<=2 keeps the next
//     chunk's two DMAs in flight while guaranteeing the current one landed.
// ---------------------------------------------------------------------------
__global__ __launch_bounds__(256)
void gemm_w_kernel(const __bf16* __restrict__ Mflat,
                   const __bf16* __restrict__ coreflat,
                   float* __restrict__ W) {
  __shared__ __align__(32) __bf16 Abuf[2][128 * KC];  // 16 KB each
  __shared__ __align__(32) __bf16 Bbuf[2][64 * KC];   //  8 KB each
  const int lane   = threadIdx.x & 31;
  const int wave   = threadIdx.x >> 5;                // 0..7
  const int mBlock = blockIdx.x * 128;                // draw rows
  const int nBase  = blockIdx.y * 64;                 // category cols
  const int nChunks = RR / KC;                        // 64

  if (wave == 0) {
    tdm_load_2d_bf16((unsigned)(size_t)&Abuf[0][0],
                     (unsigned long long)(size_t)(Mflat + (size_t)mBlock * RR),
                     KC, 128, RR, NDRAWS, RR);
    tdm_load_2d_bf16((unsigned)(size_t)&Bbuf[0][0],
                     (unsigned long long)(size_t)(coreflat + (size_t)nBase * RR),
                     KC, 64, RR, D, RR);
  }

  v8f acc[4] = {};
  for (int kc = 0; kc < nChunks; ++kc) {
    const int buf = kc & 1;
    if (wave == 0) {
      if (kc + 1 < nChunks) {
        const int nb = buf ^ 1;
        tdm_load_2d_bf16((unsigned)(size_t)&Abuf[nb][0],
                         (unsigned long long)(size_t)(Mflat + (size_t)mBlock * RR + (kc + 1) * KC),
                         KC, 128, RR, NDRAWS, RR);
        tdm_load_2d_bf16((unsigned)(size_t)&Bbuf[nb][0],
                         (unsigned long long)(size_t)(coreflat + (size_t)nBase * RR + (kc + 1) * KC),
                         KC, 64, RR, D, RR);
        __builtin_amdgcn_s_wait_tensorcnt(2);   // oldest chunk (current buf) done
      } else {
        __builtin_amdgcn_s_wait_tensorcnt(0);   // last chunk done
      }
    }
    __syncthreads();                            // DMA visible to all waves
    const __bf16* As = &Abuf[buf][wave * 16 * KC];
    const __bf16* Bs = &Bbuf[buf][0];
#pragma unroll
    for (int k0 = 0; k0 < KC; k0 += 32) {
      v16bf a = load_a_frag(As, KC, lane, k0);
#pragma unroll
      for (int nt = 0; nt < 4; ++nt) {
        v16bf bb = load_b_fragT(Bs + nt * 16 * KC, KC, lane, k0);
        acc[nt] = wmma_bf16(a, bb, acc[nt]);
      }
    }
    __syncthreads();                            // reads done before buf reuse
  }

  const int r0 = (lane & 16) ? 8 : 0;
  const int c  = lane & 15;
#pragma unroll
  for (int nt = 0; nt < 4; ++nt)
#pragma unroll
    for (int e = 0; e < 8; ++e)
      W[(size_t)(mBlock + wave * 16 + r0 + e) * D + nBase + nt * 16 + c] = acc[nt][e];
}

// ---------------------------------------------------------------------------
// K6: per-draw  M = tail_b @ head_j ; stored TRANSPOSED into Mflat
// ---------------------------------------------------------------------------
__global__ __launch_bounds__(128)
void mul_m_kernel(const __bf16* __restrict__ tail,
                  const __bf16* __restrict__ headT,
                  __bf16* __restrict__ Mflat) {
  const int b    = blockIdx.x;
  const int lane = threadIdx.x & 31;
  const int wave = threadIdx.x >> 5;
  const __bf16* A = tail + (size_t)b * RR + wave * 16 * R;
  v8f acc[4] = {};
#pragma unroll
  for (int k0 = 0; k0 < R; k0 += 32) {
    v16bf a = load_a_frag(A, R, lane, k0);
#pragma unroll
    for (int nt = 0; nt < 4; ++nt) {
      v16bf bb = load_b_fragT(headT + nt * 16 * R, R, lane, k0);
      acc[nt] = wmma_bf16(a, bb, acc[nt]);
    }
  }
  __bf16* Mout = Mflat + (size_t)b * RR;
  const int r0 = wave * 16 + ((lane & 16) ? 8 : 0);
  const int c  = lane & 15;
#pragma unroll
  for (int nt = 0; nt < 4; ++nt)
#pragma unroll
    for (int e = 0; e < 8; ++e)
      Mout[(nt * 16 + c) * R + (r0 + e)] = (__bf16)acc[nt][e];  // transposed
}

// ---------------------------------------------------------------------------
// K7: per-draw  tail_b = core_j[ix_b] @ tail_b  (in place, barrier-guarded)
// ---------------------------------------------------------------------------
__global__ __launch_bounds__(128)
void tail_update_kernel(const __bf16* __restrict__ coreflat,
                        const int* __restrict__ ix,
                        __bf16* __restrict__ tail,
                        __bf16* __restrict__ tailT) {
  const int b    = blockIdx.x;
  const int lane = threadIdx.x & 31;
  const int wave = threadIdx.x >> 5;
  const __bf16* A  = coreflat + (size_t)ix[b] * RR + wave * 16 * R;
  const __bf16* BT = tailT + (size_t)b * RR;
  v8f acc[4] = {};
#pragma unroll
  for (int k0 = 0; k0 < R; k0 += 32) {
    v16bf a = load_a_frag(A, R, lane, k0);
#pragma unroll
    for (int nt = 0; nt < 4; ++nt) {
      v16bf bb = load_b_fragT(BT + nt * 16 * R, R, lane, k0);
      acc[nt] = wmma_bf16(a, bb, acc[nt]);
    }
  }
  __syncthreads();
  __bf16* To  = tail  + (size_t)b * RR;
  __bf16* TTo = tailT + (size_t)b * RR;
  const int r0 = wave * 16 + ((lane & 16) ? 8 : 0);
  const int c  = lane & 15;
#pragma unroll
  for (int nt = 0; nt < 4; ++nt)
#pragma unroll
    for (int e = 0; e < 8; ++e) {
      float v  = acc[nt][e];
      int row  = r0 + e;
      int col  = nt * 16 + c;
      To[row * R + col]  = (__bf16)v;
      TTo[col * R + row] = (__bf16)v;
    }
}

// ---------------------------------------------------------------------------
// K8: tail init: tail_b = core7[ix_b]  (+ transposed copy)
// ---------------------------------------------------------------------------
__global__ __launch_bounds__(256)
void tail_init_kernel(const __bf16* __restrict__ coreflat,
                      const int* __restrict__ ix,
                      __bf16* __restrict__ tail,
                      __bf16* __restrict__ tailT) {
  const int b = blockIdx.x;
  const __bf16* src = coreflat + (size_t)ix[b] * RR;
  __bf16* To  = tail  + (size_t)b * RR;
  __bf16* TTo = tailT + (size_t)b * RR;
  for (int i = threadIdx.x; i < RR; i += 256) {
    __bf16 v = src[i];
    To[i] = v;
    TTo[(i & 63) * R + (i >> 6)] = v;
  }
}

// ---------------------------------------------------------------------------
// K9: Gumbel-max categorical sample. rowStride==0 => shared weight vector.
// ---------------------------------------------------------------------------
__global__ __launch_bounds__(256)
void sample_kernel(const float* __restrict__ Wt, int rowStride, int mode,
                   int* __restrict__ ix, float* __restrict__ sampleOut,
                   unsigned seed) {
  __shared__ float skey[256];
  __shared__ int   sidx[256];
  const int b = blockIdx.x;
  const int t = threadIdx.x;
  const float* row = Wt + (size_t)b * rowStride;
  float bestK = -3.4e38f;
  int   bestI = 0;
  for (int n = t; n < D; n += 256) {
    float w  = row[n];
    float lg = __logf(fmaxf(w, 1e-30f));
    unsigned h = hash3(seed, (unsigned)(b * KMODES + mode), (unsigned)n);
    float u  = ((float)(h >> 8) + 0.5f) * (1.0f / 16777216.0f);
    float g  = -__logf(-__logf(u));
    float key = lg + g;
    if (key > bestK) { bestK = key; bestI = n; }
  }
  skey[t] = bestK; sidx[t] = bestI;
  __syncthreads();
  for (int s = 128; s > 0; s >>= 1) {
    if (t < s && skey[t + s] > skey[t]) { skey[t] = skey[t + s]; sidx[t] = sidx[t + s]; }
    __syncthreads();
  }
  if (t == 0) {
    ix[b] = sidx[0];
    sampleOut[(size_t)b * KMODES + mode] = (float)sidx[0];
  }
}

// ---------------------------------------------------------------------------
// K10: log_p[b] = log(trace(tail_b)) - log(trace(norm))
// ---------------------------------------------------------------------------
__global__ __launch_bounds__(256)
void logp_kernel(const __bf16* __restrict__ tail,
                 const float* __restrict__ logTraceNorm,
                 float* __restrict__ logp) {
  int b = blockIdx.x * 256 + threadIdx.x;
  if (b >= NDRAWS) return;
  const __bf16* T = tail + (size_t)b * RR;
  float tr = 0.0f;
  for (int l = 0; l < R; ++l) tr += (float)T[l * (R + 1)];
  logp[b] = __logf(fmaxf(tr, 1e-30f)) - *logTraceNorm;
}

// ---------------------------------------------------------------------------
// Host driver
// ---------------------------------------------------------------------------
extern "C" void kernel_launch(void* const* d_in, const int* in_sizes, int n_in,
                              void* d_out, int out_size, void* d_ws, size_t ws_size,
                              hipStream_t stream) {
  (void)in_sizes; (void)n_in; (void)out_size; (void)ws_size;
  const float* cores[KMODES];
  for (int i = 0; i < KMODES; ++i) cores[i] = (const float*)d_in[i];
  float* out = (float*)d_out;             // [NDRAWS*8] sample + [NDRAWS] log_p

  char*  ws  = (char*)d_ws;
  size_t off = 0;
  auto take = [&](size_t bytes) -> void* {
    void* p = ws + off;
    off += (bytes + 255) & ~(size_t)255;
    return p;
  };
  float*  S      = (float*) take((size_t)KMODES * RR * sizeof(float));
  __bf16* headT  = (__bf16*)take((size_t)KMODES * RR * sizeof(__bf16));
  float*  ltn    = (float*) take(256);
  float*  w7     = (float*) take((size_t)D * sizeof(float));
  __bf16* coreBf = (__bf16*)take((size_t)D * RR * sizeof(__bf16));       //  8.4 MB
  __bf16* Mflat  = (__bf16*)take((size_t)NDRAWS * RR * sizeof(__bf16));  // 16.8 MB
  __bf16* tail   = (__bf16*)take((size_t)NDRAWS * RR * sizeof(__bf16));  // 16.8 MB
  __bf16* tailT  = (__bf16*)take((size_t)NDRAWS * RR * sizeof(__bf16));  // 16.8 MB
  float*  Wbuf   = (float*) take((size_t)NDRAWS * D  * sizeof(float));   //  8.4 MB
  int*    ixb    = (int*)   take((size_t)NDRAWS * sizeof(int));

  const unsigned seed = 0x5EEDu;
  const int convBlocks = (D * RR + 255) / 256;

  // Heads / normalizer
  for (int i = 0; i < KMODES; ++i)
    sum_core_kernel<<<RR / 256, 256, 0, stream>>>(cores[i], S + (size_t)i * RR);
  heads_kernel<<<1, 256, 0, stream>>>(S, headT, ltn);

  // Step j = 7: shared marginal weights, then tail = core7[ix]
  convert_core_kernel<<<convBlocks, 256, 0, stream>>>(cores[7], coreBf, D * RR);
  first_weights_kernel<<<D / 256, 256, 0, stream>>>(coreBf, headT + (size_t)7 * RR, w7);
  sample_kernel<<<NDRAWS, 256, 0, stream>>>(w7, 0, 7, ixb, out, seed);
  tail_init_kernel<<<NDRAWS, 256, 0, stream>>>(coreBf, ixb, tail, tailT);

  // Steps j = 6 .. 0
  for (int j = KMODES - 2; j >= 0; --j) {
    convert_core_kernel<<<convBlocks, 256, 0, stream>>>(cores[j], coreBf, D * RR);
    const __bf16* Aptr;
    if (j > 0) {
      mul_m_kernel<<<NDRAWS, 128, 0, stream>>>(tail, headT + (size_t)j * RR, Mflat);
      Aptr = Mflat;
    } else {
      Aptr = tailT;   // head is identity: Mflat == tail transposed == tailT
    }
    gemm_w_kernel<<<dim3(NDRAWS / 128, D / 64), 256, 0, stream>>>(Aptr, coreBf, Wbuf);
    sample_kernel<<<NDRAWS, 256, 0, stream>>>(Wbuf, D, j, ixb, out, seed);
    tail_update_kernel<<<NDRAWS, 128, 0, stream>>>(coreBf, ixb, tail, tailT);
  }

  logp_kernel<<<NDRAWS / 256, 256, 0, stream>>>(tail, ltn, out + (size_t)NDRAWS * KMODES);
}
